// MSAAttention_39118562132437
// MI455X (gfx1250) — matrix operand order repaired and checked by reading, not compiled, not run on hardware
//
#include <hip/hip_runtime.h>
#include <stdint.h>

// ---------------------------------------------------------------------------
// MSA row attention with pair bias, fused for gfx1250 (MI455X, wave32, WMMA).
// GEMMs: v_wmma_f32_16x16x32_bf16. Flash-style attention (no logits to HBM).
// A/K/V tiles staged to LDS with gfx1250 async global->LDS loads; weights are
// pre-packed once into WMMA B-fragment order (contiguous b128 loads).
// ---------------------------------------------------------------------------

#define S_DIM 128
#define R_DIM 256
#define CM    256
#define CZ    128
#define HEADS 8
#define CH    32   // head dim; == WMMA K

typedef __bf16 bf16;
typedef __attribute__((ext_vector_type(16))) __bf16 v16bf;
typedef __attribute__((ext_vector_type(8)))  float  v8f;
typedef __attribute__((ext_vector_type(4)))  int    v4i;
typedef __attribute__((ext_vector_type(2)))  int    v2i;

typedef __attribute__((address_space(1))) v4i* gv4i_p;
typedef __attribute__((address_space(3))) v4i* lv4i_p;
typedef __attribute__((address_space(1))) v2i* gv2i_p;
typedef __attribute__((address_space(3))) v2i* lv2i_p;

union BFrag { v16bf v; unsigned u[8]; bf16 h[16]; };

__device__ __forceinline__ bf16 f2bf(float x) {
  union { float f; unsigned u; } c; c.f = x;
  unsigned r = c.u + 0x7FFFu + ((c.u >> 16) & 1u);   // round-to-nearest-even
  unsigned short hv = (unsigned short)(r >> 16);
  return __builtin_bit_cast(bf16, hv);
}
__device__ __forceinline__ float bf2f(bf16 x) {
  unsigned short hv = __builtin_bit_cast(unsigned short, x);
  union { unsigned u; float f; } c; c.u = ((unsigned)hv) << 16;
  return c.f;
}

__device__ __forceinline__ v8f wmma_bf(v16bf a, v16bf b, v8f c) {
  return __builtin_amdgcn_wmma_f32_16x16x32_bf16(false, a, false, b, (short)0, c,
                                                 false, false);
}

// ------------------- gfx1250 async global->LDS copy helpers ----------------
__device__ __forceinline__ void async_copy_b128(const void* gsrc, void* ldst) {
#if __has_builtin(__builtin_amdgcn_global_load_async_to_lds_b128)
  __builtin_amdgcn_global_load_async_to_lds_b128(
      (gv4i_p)(uintptr_t)gsrc, (lv4i_p)ldst, 0, 0);
#else
  unsigned loff = (unsigned)(uintptr_t)(__attribute__((address_space(3))) void*)ldst;
  asm volatile("global_load_async_to_lds_b128 %0, %1, off"
               :: "v"(loff), "v"((uint64_t)(uintptr_t)gsrc) : "memory");
#endif
}
__device__ __forceinline__ void async_copy_b64(const void* gsrc, void* ldst) {
#if __has_builtin(__builtin_amdgcn_global_load_async_to_lds_b64)
  __builtin_amdgcn_global_load_async_to_lds_b64(
      (gv2i_p)(uintptr_t)gsrc, (lv2i_p)ldst, 0, 0);
#else
  unsigned loff = (unsigned)(uintptr_t)(__attribute__((address_space(3))) void*)ldst;
  asm volatile("global_load_async_to_lds_b64 %0, %1, off"
               :: "v"(loff), "v"((uint64_t)(uintptr_t)gsrc) : "memory");
#endif
}
__device__ __forceinline__ void wait_async0() {
#if __has_builtin(__builtin_amdgcn_s_wait_asynccnt)
  __builtin_amdgcn_s_wait_asynccnt(0);
#else
  asm volatile("s_wait_asynccnt 0" ::: "memory");
#endif
}

// A fragment 16x32 bf16 from an LDS tile with leading dim 32.
// Contiguous k order used here (not AMD's table): lane = M%16; VGPR v holds K-pair: K = 16*half + 2v.
__device__ __forceinline__ v16bf load_a_lds(const bf16* t, int row0, int l16,
                                            int half) {
  BFrag f;
#pragma unroll
  for (int i = 0; i < 8; ++i) {
    int kk = 16 * half + 2 * i;
    f.u[i] = *(const unsigned*)(t + (row0 + l16) * 32 + kk);
  }
  return f.v;
}

// A fragment 16x32 bf16 directly from row-major global memory.
__device__ __forceinline__ v16bf load_a_rm(const bf16* base, int row0, int k0,
                                           int ld, int l16, int half) {
  BFrag f;
#pragma unroll
  for (int i = 0; i < 8; ++i) {
    int kk = 16 * half + 2 * i;
    f.u[i] = *(const unsigned*)(base + (size_t)(row0 + l16) * ld + k0 + kk);
  }
  return f.v;
}

// B fragment 32x16 where B[K][N] = mem[N][K] (K contiguous): used for K^T.
__device__ __forceinline__ v16bf load_b_trans(const bf16* base, int n0, int k0,
                                              int ld, int l16, int half) {
  BFrag f;
#pragma unroll
  for (int i = 0; i < 8; ++i) {
    int kk = 16 * half + 2 * i;
    f.u[i] = *(const unsigned*)(base + (size_t)(n0 + l16) * ld + k0 + kk);
  }
  return f.v;
}

// B fragment 32x16 from row-major [K][N] bf16 (strided K pairs).
__device__ __forceinline__ v16bf load_b_rm_bf16(const bf16* base, int k0, int n0,
                                                int ld, int l16, int half) {
  BFrag f;
#pragma unroll
  for (int i = 0; i < 8; ++i) {
    int kk = 16 * half + 2 * i;
    f.h[2 * i]     = base[(size_t)(k0 + kk) * ld + n0 + l16];
    f.h[2 * i + 1] = base[(size_t)(k0 + kk + 1) * ld + n0 + l16];
  }
  return f.v;
}

// B fragment from pre-packed fragment-order memory: 8 contiguous dwords/lane.
__device__ __forceinline__ v16bf load_b_packed(const unsigned* p) {
  BFrag f;
#pragma unroll
  for (int i = 0; i < 8; ++i) f.u[i] = p[i];
  return f.v;
}

__device__ __forceinline__ float wave_sum32(float x) {
#pragma unroll
  for (int m = 1; m < 32; m <<= 1) x += __shfl_xor(x, m, 32);
  return x;
}

// ------------------------------ LayerNorm(m) -> bf16 -----------------------
__global__ __launch_bounds__(256) void msa_ln_m(const float* __restrict__ m,
                                                const float* __restrict__ g,
                                                const float* __restrict__ b,
                                                bf16* __restrict__ out) {
  int wave = threadIdx.x >> 5, lane = threadIdx.x & 31;
  size_t row = (size_t)blockIdx.x * 8 + wave;           // S*R rows of width 256
  const float* p = m + row * CM + lane * 8;
  float x[8]; float s = 0.f;
#pragma unroll
  for (int i = 0; i < 8; ++i) { x[i] = p[i]; s += x[i]; }
  s = wave_sum32(s);
  float mu = s * (1.f / CM);
  float vs = 0.f;
#pragma unroll
  for (int i = 0; i < 8; ++i) { float d = x[i] - mu; vs += d * d; }
  vs = wave_sum32(vs);
  float inv = rsqrtf(vs * (1.f / CM) + 1e-5f);
  bf16* o = out + row * CM + lane * 8;
#pragma unroll
  for (int i = 0; i < 8; ++i) {
    int c = lane * 8 + i;
    o[i] = f2bf((x[i] - mu) * inv * g[c] + b[c]);
  }
}

// ------------------- LayerNorm(z) @ w_z -> zb[H][R][R] f32 -----------------
__global__ __launch_bounds__(256) void msa_ln_z(const float* __restrict__ z,
                                                const float* __restrict__ g,
                                                const float* __restrict__ b,
                                                const float* __restrict__ wz,
                                                float* __restrict__ zb) {
  int wave = threadIdx.x >> 5, lane = threadIdx.x & 31;
  size_t row = (size_t)blockIdx.x * 8 + wave;           // R*R rows of width 128
  const float* p = z + row * CZ + lane * 4;
  float x[4]; float s = 0.f;
#pragma unroll
  for (int i = 0; i < 4; ++i) { x[i] = p[i]; s += x[i]; }
  s = wave_sum32(s);
  float mu = s * (1.f / CZ);
  float vs = 0.f;
#pragma unroll
  for (int i = 0; i < 4; ++i) { float d = x[i] - mu; vs += d * d; }
  vs = wave_sum32(vs);
  float inv = rsqrtf(vs * (1.f / CZ) + 1e-5f);
  float acc[HEADS];
#pragma unroll
  for (int h = 0; h < HEADS; ++h) acc[h] = 0.f;
#pragma unroll
  for (int i = 0; i < 4; ++i) {
    int c = lane * 4 + i;
    float zn = (x[i] - mu) * inv * g[c] + b[c];
#pragma unroll
    for (int h = 0; h < HEADS; ++h) acc[h] += zn * wz[c * HEADS + h];
  }
#pragma unroll
  for (int h = 0; h < HEADS; ++h) acc[h] = wave_sum32(acc[h]);
  if (lane == 0) {
#pragma unroll
    for (int h = 0; h < HEADS; ++h)
      zb[(size_t)h * (R_DIM * R_DIM) + row] = acc[h];
  }
}

// -------- one-time weight pack: f32 [256,256] -> bf16 B-fragment order -----
// Fragment (ks, nt) = 32-K step x 16-N tile; stored as [frag][lane][8 dwords].
// 128 fragments * 1KB = 32768 dwords per weight; grid.z selects the weight.
__global__ __launch_bounds__(256) void msa_wpack(const float* __restrict__ wq,
                                                 const float* __restrict__ wk,
                                                 const float* __restrict__ wv,
                                                 const float* __restrict__ wg,
                                                 const float* __restrict__ wo,
                                                 unsigned* __restrict__ out) {
  int widx = blockIdx.z;
  const float* W = (widx == 0) ? wq : (widx == 1) ? wk : (widx == 2) ? wv
                 : (widx == 3) ? wg : wo;
  unsigned* op = out + (size_t)widx * 32768;
  int wave = threadIdx.x >> 5, lane = threadIdx.x & 31;
  int l16 = lane & 15, half = lane >> 4;
  int frag = blockIdx.x * 8 + wave;          // 0..127
  int k0 = (frag >> 4) * 32, n0 = (frag & 15) * 16;
  unsigned* dst = op + (size_t)frag * 256 + lane * 8;
#pragma unroll
  for (int i = 0; i < 8; ++i) {
    int kk = 16 * half + 2 * i;
    unsigned short lo = __builtin_bit_cast(unsigned short,
        f2bf(W[(size_t)(k0 + kk) * CM + n0 + l16]));
    unsigned short hi = __builtin_bit_cast(unsigned short,
        f2bf(W[(size_t)(k0 + kk + 1) * CM + n0 + l16]));
    dst[i] = ((unsigned)hi << 16) | lo;
  }
}

// ------------- fused Q/K/V/G projections: [S*R,256] @ [256,256] ------------
// block tile 128x64 (8 waves of 32x32). A tile async-staged to LDS; weights
// read from packed fragments. grid.z selects the projection.
__global__ __launch_bounds__(256) void msa_proj(const bf16* __restrict__ A,
                                                const unsigned* __restrict__ wpack,
                                                const float* __restrict__ bg,
                                                bf16* __restrict__ oq,
                                                bf16* __restrict__ ok,
                                                bf16* __restrict__ ov,
                                                bf16* __restrict__ og) {
  __shared__ bf16 atile[128 * 32];           // 8 KB
  int mode = blockIdx.z;
  bf16* out = (mode == 0) ? oq : (mode == 1) ? ok : (mode == 2) ? ov : og;
  const unsigned* wp = wpack + (size_t)mode * 32768;

  int tid = threadIdx.x;
  int wave = tid >> 5, lane = tid & 31;
  int l16 = lane & 15, half = lane >> 4;
  int wm = wave >> 1, wn = wave & 1;
  int m0b = blockIdx.x * 128;
  int m0 = m0b + wm * 32;
  int n0 = blockIdx.y * 64 + wn * 32;
  int ntf = blockIdx.y * 4 + wn * 2;         // fragment N index (of 16)

  v8f acc[2][2] = {};
  for (int k0 = 0; k0 < CM; k0 += 32) {
    // stage A tile 128x32 bf16 (8KB): 512 x 16B chunks, 2 per thread, async
#pragma unroll
    for (int c = 0; c < 2; ++c) {
      int chunk = tid * 2 + c;               // 0..511
      int row = chunk >> 2, part = chunk & 3;
      async_copy_b128(A + (size_t)(m0b + row) * CM + k0 + part * 8,
                      atile + row * 32 + part * 8);
    }
    wait_async0();
    __syncthreads();

    v16bf a0 = load_a_lds(atile, wm * 32,      l16, half);
    v16bf a1 = load_a_lds(atile, wm * 32 + 16, l16, half);
    int ks = k0 >> 5;
    v16bf b0 = load_b_packed(wp + ((size_t)(ks * 16 + ntf    )) * 256 + lane * 8);
    v16bf b1 = load_b_packed(wp + ((size_t)(ks * 16 + ntf + 1)) * 256 + lane * 8);
    acc[0][0] = wmma_bf(a0, b0, acc[0][0]);
    acc[0][1] = wmma_bf(a0, b1, acc[0][1]);
    acc[1][0] = wmma_bf(a1, b0, acc[1][0]);
    acc[1][1] = wmma_bf(a1, b1, acc[1][1]);
    __syncthreads();                          // before next stage overwrites
  }

  const float qscale = 0.17677669529663687f;  // 1/sqrt(32)
#pragma unroll
  for (int i = 0; i < 2; ++i)
#pragma unroll
    for (int j = 0; j < 2; ++j)
#pragma unroll
      for (int r = 0; r < 8; ++r) {
        int row = m0 + i * 16 + r + 8 * half;
        int col = n0 + j * 16 + l16;
        float val = acc[i][j][r];
        if (mode == 0) val *= qscale;
        else if (mode == 3)
          val = __builtin_amdgcn_rcpf(1.f + __expf(-(val + bg[col])));
        out[(size_t)row * CM + col] = f2bf(val);
      }
}

// ---------------- fused attention per (s,h): flash-style -------------------
// 16 waves; wave w owns query rows [16w,16w+16). K/V 32-key blocks are
// async-staged to LDS once per block; online softmax; P re-layouts via LDS.
__global__ __launch_bounds__(512) void msa_attn(const bf16* __restrict__ q,
                                                const bf16* __restrict__ k,
                                                const bf16* __restrict__ v,
                                                const bf16* __restrict__ gate,
                                                const float* __restrict__ mask,
                                                const float* __restrict__ zb,
                                                bf16* __restrict__ og) {
  __shared__ bf16 kt[32 * 32];                // 2 KB: [key][c]
  __shared__ bf16 vt[32 * 32];                // 2 KB: [key][c]
  __shared__ bf16 lds_p[16][16 * 32];         // per-wave 16x32 P tile

  int s = blockIdx.x >> 3, h = blockIdx.x & 7;
  int tid = threadIdx.x;
  int wave = tid >> 5, lane = tid & 31;
  int l16 = lane & 15, half = lane >> 4;
  int q0 = wave * 16;

  const bf16* qb = q + (size_t)s * R_DIM * CM;
  const bf16* kb = k + (size_t)s * R_DIM * CM;
  const bf16* vb = v + (size_t)s * R_DIM * CM;
  const float* zrow = zb + (size_t)h * (R_DIM * R_DIM);

  v16bf aq = load_a_rm(qb, q0, h * CH, CM, l16, half);   // 16 x 32 (K = CH)
  v8f o0 = {}, o1 = {};
  float Mr[8], Lr[8];
#pragma unroll
  for (int r = 0; r < 8; ++r) { Mr[r] = -1e30f; Lr[r] = 0.f; }

  bf16* lp = &lds_p[wave][0];

  // per-thread staging chunk: 512 threads x 8B cover K(2KB)+V(2KB)
  int sel  = tid >> 8;                        // 0: K, 1: V
  int idx  = tid & 255;
  int skey = idx >> 3, spart = idx & 7;       // 8B = 4 bf16 per part
  const bf16* sbase = sel ? vb : kb;
  bf16* sdst = (sel ? vt : kt) + skey * 32 + spart * 4;

  for (int kb0 = 0; kb0 < R_DIM; kb0 += 32) {
    async_copy_b64(sbase + (size_t)(kb0 + skey) * CM + h * CH + spart * 4, sdst);
    wait_async0();
    __syncthreads();

    // logits = Q . K^T for 16 q rows x 32 keys (two 16-col WMMA tiles)
    v16bf bk0 = load_b_trans(kt, 0,  0, 32, l16, half);
    v16bf bk1 = load_b_trans(kt, 16, 0, 32, l16, half);
    v8f zc = {};
    v8f c0 = wmma_bf(aq, bk0, zc);
    v8f c1 = wmma_bf(aq, bk1, zc);

    float mb0 = 1e9f * (mask[(size_t)s * R_DIM + kb0 + l16] - 1.f);
    float mb1 = 1e9f * (mask[(size_t)s * R_DIM + kb0 + 16 + l16] - 1.f);

#pragma unroll
    for (int r = 0; r < 8; ++r) {
      int row = q0 + r + 8 * half;
      float t0 = c0[r] + mb0 + zrow[(size_t)row * R_DIM + kb0 + l16];
      float t1 = c1[r] + mb1 + zrow[(size_t)row * R_DIM + kb0 + 16 + l16];
      float bm = fmaxf(t0, t1);
      bm = fmaxf(bm, __shfl_xor(bm, 1, 32));
      bm = fmaxf(bm, __shfl_xor(bm, 2, 32));
      bm = fmaxf(bm, __shfl_xor(bm, 4, 32));
      bm = fmaxf(bm, __shfl_xor(bm, 8, 32));
      float Mn = fmaxf(Mr[r], bm);
      float alpha = __expf(Mr[r] - Mn);
      float p0 = __expf(t0 - Mn);
      float p1 = __expf(t1 - Mn);
      float rs = p0 + p1;
      rs += __shfl_xor(rs, 1, 32);
      rs += __shfl_xor(rs, 2, 32);
      rs += __shfl_xor(rs, 4, 32);
      rs += __shfl_xor(rs, 8, 32);
      Lr[r] = Lr[r] * alpha + rs;
      Mr[r] = Mn;
      o0[r] *= alpha;
      o1[r] *= alpha;
      int rl = r + 8 * half;
      lp[rl * 32 + l16]      = f2bf(p0);
      lp[rl * 32 + 16 + l16] = f2bf(p1);
    }
    asm volatile("s_wait_dscnt 0" ::: "memory");

    // reload probabilities as A fragment (16 q rows x 32 keys)
    BFrag ap;
#pragma unroll
    for (int i = 0; i < 8; ++i) {
      int kk = 16 * half + 2 * i;
      ap.u[i] = *(const unsigned*)&lp[l16 * 32 + kk];
    }
    // O += P . V  (V block from LDS: 32 keys x 32 channels)
    v16bf bv0 = load_b_rm_bf16(vt, 0, 0,  32, l16, half);
    v16bf bv1 = load_b_rm_bf16(vt, 0, 16, 32, l16, half);
    o0 = wmma_bf(ap.v, bv0, o0);
    o1 = wmma_bf(ap.v, bv1, o1);
    __syncthreads();                          // kt/vt reused next iteration
  }

  // epilogue: normalize, apply gate, store bf16 [S,R,H*C]
#pragma unroll
  for (int r = 0; r < 8; ++r) {
    int row = q0 + r + 8 * half;
    float inv = __builtin_amdgcn_rcpf(Lr[r]);
    size_t base = ((size_t)s * R_DIM + row) * CM + h * CH;
    float g0 = bf2f(gate[base + l16]);
    float g1 = bf2f(gate[base + 16 + l16]);
    og[base + l16]      = f2bf(o0[r] * inv * g0);
    og[base + 16 + l16] = f2bf(o1[r] * inv * g1);
  }
}

// ------------------- output projection: og @ wo + bo -> f32 ----------------
__global__ __launch_bounds__(256) void msa_outproj(const bf16* __restrict__ A,
                                                   const unsigned* __restrict__ wopack,
                                                   const float* __restrict__ bo,
                                                   float* __restrict__ out) {
  __shared__ bf16 atile[128 * 32];
  int tid = threadIdx.x;
  int wave = tid >> 5, lane = tid & 31;
  int l16 = lane & 15, half = lane >> 4;
  int wm = wave >> 1, wn = wave & 1;
  int m0b = blockIdx.x * 128;
  int m0 = m0b + wm * 32;
  int n0 = blockIdx.y * 64 + wn * 32;
  int ntf = blockIdx.y * 4 + wn * 2;

  v8f acc[2][2] = {};
  for (int k0 = 0; k0 < CM; k0 += 32) {
#pragma unroll
    for (int c = 0; c < 2; ++c) {
      int chunk = tid * 2 + c;
      int row = chunk >> 2, part = chunk & 3;
      async_copy_b128(A + (size_t)(m0b + row) * CM + k0 + part * 8,
                      atile + row * 32 + part * 8);
    }
    wait_async0();
    __syncthreads();

    v16bf a0 = load_a_lds(atile, wm * 32,      l16, half);
    v16bf a1 = load_a_lds(atile, wm * 32 + 16, l16, half);
    int ks = k0 >> 5;
    v16bf b0 = load_b_packed(wopack + ((size_t)(ks * 16 + ntf    )) * 256 + lane * 8);
    v16bf b1 = load_b_packed(wopack + ((size_t)(ks * 16 + ntf + 1)) * 256 + lane * 8);
    acc[0][0] = wmma_bf(a0, b0, acc[0][0]);
    acc[0][1] = wmma_bf(a0, b1, acc[0][1]);
    acc[1][0] = wmma_bf(a1, b0, acc[1][0]);
    acc[1][1] = wmma_bf(a1, b1, acc[1][1]);
    __syncthreads();
  }
#pragma unroll
  for (int i = 0; i < 2; ++i)
#pragma unroll
    for (int j = 0; j < 2; ++j)
#pragma unroll
      for (int r = 0; r < 8; ++r) {
        int row = m0 + i * 16 + r + 8 * half;
        int col = n0 + j * 16 + l16;
        out[(size_t)row * CM + col] = acc[i][j][r] + bo[col];
      }
}

// ---------------------------------------------------------------------------
extern "C" void kernel_launch(void* const* d_in, const int* in_sizes, int n_in,
                              void* d_out, int out_size, void* d_ws, size_t ws_size,
                              hipStream_t stream) {
  (void)in_sizes; (void)n_in; (void)out_size; (void)ws_size;
  const float* m      = (const float*)d_in[0];
  const float* z      = (const float*)d_in[1];
  const float* mask   = (const float*)d_in[2];
  const float* ln_m_g = (const float*)d_in[3];
  const float* ln_m_b = (const float*)d_in[4];
  const float* ln_z_g = (const float*)d_in[5];
  const float* ln_z_b = (const float*)d_in[6];
  const float* w_z    = (const float*)d_in[7];
  const float* wq     = (const float*)d_in[8];
  const float* wk     = (const float*)d_in[9];
  const float* wv     = (const float*)d_in[10];
  const float* wg     = (const float*)d_in[11];
  const float* bg     = (const float*)d_in[12];
  const float* wo     = (const float*)d_in[13];
  const float* bo     = (const float*)d_in[14];
  float* out = (float*)d_out;

  const size_t SR  = (size_t)S_DIM * R_DIM;        // 32768
  const size_t BUF = SR * CM * sizeof(bf16);       // 16.78 MB per bf16 tensor
  char* ws = (char*)d_ws;
  bf16* m_bf  = (bf16*)(ws + 0 * BUF);
  bf16* q_bf  = (bf16*)(ws + 1 * BUF);
  bf16* k_bf  = (bf16*)(ws + 2 * BUF);
  bf16* v_bf  = (bf16*)(ws + 3 * BUF);
  bf16* g_bf  = (bf16*)(ws + 4 * BUF);
  bf16* og_bf = (bf16*)(ws + 5 * BUF);
  float*    zb    = (float*)(ws + 6 * BUF);        // H*R*R f32 = 2 MB
  unsigned* wpack = (unsigned*)(ws + 6 * BUF + (size_t)HEADS * R_DIM * R_DIM * 4);

  // 0) pack all five weight matrices into WMMA B-fragment order (bf16)
  msa_wpack<<<dim3(16, 1, 5), dim3(256), 0, stream>>>(wq, wk, wv, wg, wo, wpack);
  // 1) LayerNorm(m) -> bf16
  msa_ln_m<<<dim3(SR / 8), dim3(256), 0, stream>>>(m, ln_m_g, ln_m_b, m_bf);
  // 2) LayerNorm(z) @ w_z -> zb[H][R][R]
  msa_ln_z<<<dim3((R_DIM * R_DIM) / 8), dim3(256), 0, stream>>>(z, ln_z_g, ln_z_b, w_z, zb);
  // 3) q/k/v/gate projections (WMMA, async LDS staging)
  msa_proj<<<dim3(SR / 128, CM / 64, 4), dim3(256), 0, stream>>>(
      m_bf, wpack, bg, q_bf, k_bf, v_bf, g_bf);
  // 4) fused attention (WMMA + online softmax + async K/V staging)
  msa_attn<<<dim3(S_DIM * HEADS), dim3(512), 0, stream>>>(
      q_bf, k_bf, v_bf, g_bf, mask, zb, og_bf);
  // 5) output projection (WMMA) -> f32
  msa_outproj<<<dim3(SR / 128, CM / 64, 1), dim3(256), 0, stream>>>(
      og_bf, wpack + 4 * 32768, bo, out);
}
